// KQV_Attention_1812476198974
// MI455X (gfx1250) — compile-verified, matrix operand order checked by reading, not run on hardware
//
#include <hip/hip_runtime.h>
#include <hip/hip_bf16.h>

typedef __attribute__((ext_vector_type(16))) __bf16 v16bf;
typedef __attribute__((ext_vector_type(8)))  float  v8f;

#define B_    64
#define L_    2048
#define DTS   8
#define DSEQ  120
#define DIN   128
#define H_    64
#define SEG   4
#define TOK_PER_SEG   (L_ / SEG)        /* 512  */
#define TILES_PER_SEG (TOK_PER_SEG/16)  /* 32   */
#define SLOPE 0.2291666666666667f       /* (1/8 + 1/3)/2 */
#define PART_STRIDE 66                  /* m, Z, acc[64] */

__device__ __forceinline__ float rrelu_f(float x) { return x >= 0.f ? x : x * SLOPE; }

// Fused: x = concat(ts,seq); K/V via bf16 WMMA; streaming masked softmax (per-segment partial).
__global__ __launch_bounds__(256)
void attn_partial_kernel(const float* __restrict__ ts,
                         const float* __restrict__ seq,
                         const int*   __restrict__ lengths,
                         const float* __restrict__ Wk,
                         const float* __restrict__ bk,
                         const float* __restrict__ Wq,
                         const float* __restrict__ bq,
                         const float* __restrict__ Wv,
                         const float* __restrict__ bv,
                         float* __restrict__ part) {
  // B-operand layout [mat][kchunk][ntile][lane][16 bf16] (32B per lane slot)
  __shared__ __align__(32) __bf16 wlds[2][4][4][32][16];
  __shared__ float q_s[H_];
  __shared__ float red_m[8], red_z[8], red_acc[8][H_];

  const int b    = blockIdx.x / SEG;
  const int seg  = blockIdx.x % SEG;
  const int t    = threadIdx.x;
  const int lane = t & 31;
  const int wave = t >> 5;
  const int len  = lengths[b];

  // ---- pack Wk / Wv (fp32 [128][64]) into WMMA B-operand bf16 lane layout ----
  // element e of a lane: pair=e>>1, odd=e&1, K = (pair<4 ? 2*pair : 16+2*(pair-4)) + (lane>=16 ? 8:0) + odd
  for (int i = t; i < 2 * 8192; i += 256) {
    const int m  = i >> 13;
    const int r  = i & 8191;
    const int e  = r & 15;
    const int ln = (r >> 4) & 31;
    const int n  = (r >> 9) & 3;
    const int c  = (r >> 11) & 3;
    const int pair = e >> 1, odd = e & 1;
    const int kb = (pair < 4) ? (2 * pair) : (16 + 2 * (pair - 4));
    const int d  = 32 * c + kb + ((ln >> 4) ? 8 : 0) + odd;
    const int h  = n * 16 + (ln & 15);
    const float* W = m ? Wv : Wk;
    wlds[m][c][n][ln][e] = (__bf16)W[d * H_ + h];
  }

  // ---- q_last for this batch (tiny: 128-dot per h) ----
  if (t < H_) {
    const int pos  = len - 1;
    const int rowt = b * L_ + pos;
    float s = bq[t];
    #pragma unroll 4
    for (int d = 0; d < DIN; ++d) {
      const float xd = (d < DTS) ? ts[rowt * DTS + d] : seq[rowt * DSEQ + (d - DTS)];
      s += xd * Wq[d * H_ + t];
    }
    q_s[t] = rrelu_f(s);
  }
  __syncthreads();

  const int nloc = lane & 15;
  const int half = lane >> 4;
  float qr[4], bkr[4], bvr[4];
  #pragma unroll
  for (int n = 0; n < 4; ++n) {
    const int h = n * 16 + nloc;
    qr[n] = q_s[h]; bkr[n] = bk[h]; bvr[n] = bv[h];
  }

  const v8f z8 = {0.f,0.f,0.f,0.f,0.f,0.f,0.f,0.f};
  float m_run = -__builtin_inff();
  float z_run = 0.f;
  float acc[4] = {0.f, 0.f, 0.f, 0.f};

  for (int tile = wave; tile < TILES_PER_SEG; tile += 8) {
    const int tbase = seg * TOK_PER_SEG + tile * 16;
    const int token = tbase + nloc;           // A-layout: lane&15 == row M
    const int row   = b * L_ + token;

    v8f ck[4] = { z8, z8, z8, z8 };
    v8f cv[4] = { z8, z8, z8, z8 };

    #pragma unroll
    for (int c = 0; c < 4; ++c) {
      // Per-lane A operand: features K = 32c + {run0: 0..7 | 8..15 (half)}, {run1: +16}
      const int d0 = 32 * c + (half ? 8 : 0);
      const float* p0 = (d0 == 0) ? (ts + row * DTS)
                                  : (seq + row * DSEQ + (d0 - DTS));
      const float* p1 = seq + row * DSEQ + (d0 + 16 - DTS);
      const float4 f0 = *(const float4*)(p0);
      const float4 f1 = *(const float4*)(p0 + 4);
      const float4 f2 = *(const float4*)(p1);
      const float4 f3 = *(const float4*)(p1 + 4);
      union { __bf16 arr[16]; v16bf vec; } ua;
      ua.arr[0]=(__bf16)f0.x;  ua.arr[1]=(__bf16)f0.y;  ua.arr[2]=(__bf16)f0.z;  ua.arr[3]=(__bf16)f0.w;
      ua.arr[4]=(__bf16)f1.x;  ua.arr[5]=(__bf16)f1.y;  ua.arr[6]=(__bf16)f1.z;  ua.arr[7]=(__bf16)f1.w;
      ua.arr[8]=(__bf16)f2.x;  ua.arr[9]=(__bf16)f2.y;  ua.arr[10]=(__bf16)f2.z; ua.arr[11]=(__bf16)f2.w;
      ua.arr[12]=(__bf16)f3.x; ua.arr[13]=(__bf16)f3.y; ua.arr[14]=(__bf16)f3.z; ua.arr[15]=(__bf16)f3.w;
      const v16bf a = ua.vec;

      #pragma unroll
      for (int n = 0; n < 4; ++n) {
        const v16bf bkm = *(const v16bf*)&wlds[0][c][n][lane][0];
        ck[n] = __builtin_amdgcn_wmma_f32_16x16x32_bf16(false, a, false, bkm, (short)0, ck[n], false, false);
        const v16bf bvm = *(const v16bf*)&wlds[1][c][n][lane][0];
        cv[n] = __builtin_amdgcn_wmma_f32_16x16x32_bf16(false, a, false, bvm, (short)0, cv[n], false, false);
      }
    }

    // scores: lane-partial over its h slice, rows match C layout (r -> token tbase + r + 8*half)
    float sc[8];
    #pragma unroll
    for (int r = 0; r < 8; ++r) {
      float s = 0.f;
      #pragma unroll
      for (int n = 0; n < 4; ++n)
        s += qr[n] * rrelu_f(ck[n][r] + bkr[n]);
      sc[r] = s;
    }
    #pragma unroll
    for (int r = 0; r < 8; ++r) {
      #pragma unroll
      for (int mm = 1; mm < 16; mm <<= 1)
        sc[r] += __shfl_xor(sc[r], mm, 32);
    }

    // masking + online softmax update
    const int rowbase = tbase + (half ? 8 : 0);
    bool vld[8];
    float tmax = -__builtin_inff();
    #pragma unroll
    for (int r = 0; r < 8; ++r) {
      vld[r] = (rowbase + r) < len;
      if (vld[r]) tmax = fmaxf(tmax, sc[r]);
    }
    tmax = fmaxf(tmax, __shfl_xor(tmax, 16, 32));
    const float mn = fmaxf(m_run, tmax);
    if (mn != -__builtin_inff()) {
      const float scale = __expf(m_run - mn);   // 0 when m_run == -inf
      float p[8], lsum = 0.f;
      #pragma unroll
      for (int r = 0; r < 8; ++r) {
        p[r] = vld[r] ? __expf(sc[r] - mn) : 0.f;
        lsum += p[r];
      }
      lsum += __shfl_xor(lsum, 16, 32);
      z_run = z_run * scale + lsum;
      #pragma unroll
      for (int n = 0; n < 4; ++n) {
        float a2 = 0.f;
        #pragma unroll
        for (int r = 0; r < 8; ++r)
          a2 += p[r] * rrelu_f(cv[n][r] + bvr[n]);
        acc[n] = acc[n] * scale + a2;
      }
      m_run = mn;
    }
  }

  // fold the two token-halves (lane and lane^16 hold same h, disjoint tokens)
  #pragma unroll
  for (int n = 0; n < 4; ++n) acc[n] += __shfl_xor(acc[n], 16, 32);
  if (lane < 16) {
    #pragma unroll
    for (int n = 0; n < 4; ++n) red_acc[wave][n * 16 + lane] = acc[n];
  }
  if (lane == 0) { red_m[wave] = m_run; red_z[wave] = z_run; }
  __syncthreads();

  // combine 8 waves -> segment partial (m*, Z, acc[64]) in workspace
  if (t < H_) {
    float mstar = -__builtin_inff();
    #pragma unroll
    for (int w = 0; w < 8; ++w) mstar = fmaxf(mstar, red_m[w]);
    float num = 0.f, den = 0.f;
    #pragma unroll
    for (int w = 0; w < 8; ++w) {
      const float mw = red_m[w];
      const float e = (mw == -__builtin_inff()) ? 0.f : __expf(mw - mstar);
      den += e * red_z[w];
      num += e * red_acc[w][t];
    }
    float* pp = part + (size_t)(b * SEG + seg) * PART_STRIDE;
    if (t == 0) { pp[0] = mstar; pp[1] = den; }
    pp[2 + t] = num;
  }
}

// log-sum-exp combine of the SEG partials per batch
__global__ __launch_bounds__(64)
void attn_reduce_kernel(const float* __restrict__ part, float* __restrict__ out) {
  const int b = blockIdx.x;
  const int h = threadIdx.x;
  float mstar = -__builtin_inff();
  for (int s = 0; s < SEG; ++s)
    mstar = fmaxf(mstar, part[(size_t)(b * SEG + s) * PART_STRIDE]);
  float num = 0.f, den = 0.f;
  for (int s = 0; s < SEG; ++s) {
    const float* pp = part + (size_t)(b * SEG + s) * PART_STRIDE;
    const float mw = pp[0];
    const float e = (mw == -__builtin_inff()) ? 0.f : __expf(mw - mstar);
    den += e * pp[1];
    num += e * pp[2 + h];
  }
  out[b * H_ + h] = num / den;   // every batch has >=1 valid token
}

extern "C" void kernel_launch(void* const* d_in, const int* in_sizes, int n_in,
                              void* d_out, int out_size, void* d_ws, size_t ws_size,
                              hipStream_t stream) {
  const float* ts  = (const float*)d_in[0];
  const float* seq = (const float*)d_in[1];
  const int*   len = (const int*)  d_in[2];
  const float* Wk  = (const float*)d_in[3];
  const float* bk  = (const float*)d_in[4];
  const float* Wq  = (const float*)d_in[5];
  const float* bq  = (const float*)d_in[6];
  const float* Wv  = (const float*)d_in[7];
  const float* bv  = (const float*)d_in[8];
  float* part = (float*)d_ws;   // B*SEG*66 floats = 66 KB
  float* out  = (float*)d_out;

  attn_partial_kernel<<<B_ * SEG, 256, 0, stream>>>(ts, seq, len, Wk, bk, Wq, bq, Wv, bv, part);
  attn_reduce_kernel<<<B_, 64, 0, stream>>>(part, out);
}
